// SwinTransformerBlock_2989297238088
// MI455X (gfx1250) — matrix operand order repaired and checked by reading, not compiled, not run on hardware
//
#include <hip/hip_runtime.h>
#include <math.h>

// ---------------------------------------------------------------------------
// Swin block for MI455X (gfx1250, wave32). All heavy matmuls run through
// v_wmma_f32_16x16x32_bf16. fp32 activations are converted to bf16 on the fly.
// ---------------------------------------------------------------------------

typedef __bf16 bf16;
typedef __attribute__((ext_vector_type(16))) __bf16 v16bf;
typedef __attribute__((ext_vector_type(8)))  float   v8f;

__device__ __forceinline__ v8f wmma_bf16(v16bf a, v16bf b, v8f c) {
  return __builtin_amdgcn_wmma_f32_16x16x32_bf16(false, a, false, b, (short)0, c,
                                                 false, false);
}

// A-fragment: 16x32 bf16. lane<16 holds K={kb..kb+7, kb+16..kb+23} with kb=0,
// lane>=16 with kb=8 (ISA 7.12.2 "16-bit A-Matrix 16x32").
__device__ __forceinline__ v16bf frag2(const bf16* p0, const bf16* p1) {
  v16bf r;
#pragma unroll
  for (int i = 0; i < 8; ++i) { r[i] = p0[i]; r[i + 8] = p1[i]; }
  return r;
}
// B-fragment: 32x16 bf16, in this code's k order (A uses it too). lane holds column n=lane&15, 16 contiguous K values
// starting at kb=(lane<16?0:16) when stored N-major (B^T row-major).
__device__ __forceinline__ v16bf frag16(const bf16* p) {
  v16bf r;
#pragma unroll
  for (int i = 0; i < 16; ++i) r[i] = p[i];
  return r;
}

__device__ __forceinline__ unsigned pack2(bf16 a, bf16 b) {
  union { bf16 h[2]; unsigned u; } p;
  p.h[0] = a; p.h[1] = b;
  return p.u;
}

// ---------------------------------------------------------------------------
// Generic bf16 GEMM: C = A(MxK) * B(KxN) + bias, tile 128x128x32, 256 threads
// (8 waves), each wave owns a 16-row strip across all 128 columns (8 WMMAs
// per K-step reusing one A fragment). Epilogues are templated:
//   EPI 0: bf16 store (QKV)
//   EPI 1: window-reverse + roll + shortcut residual scatter, fp32 (proj)
//   EPI 2: tanh-GELU -> bf16 (MLP fc1)
//   EPI 3: + residual -> fp32 (MLP fc2, in-place on d_out)
// ---------------------------------------------------------------------------
template <int EPI>
__global__ void gemm_bf16_kernel(const bf16* __restrict__ A,
                                 const bf16* __restrict__ Bw,
                                 const float* __restrict__ bias,
                                 int M, int N, int K,
                                 float* outF, bf16* outB, int ldo,
                                 const float* resF,
                                 int ws, int ss, int cbase) {
  __shared__ bf16 As[128 * 32];  // row-major [m][k]   (8 KB)
  __shared__ bf16 Bs[128 * 32];  // B^T: [n][k]        (8 KB)
  const int tid  = threadIdx.x;
  const int lane = tid & 31;
  const int wv   = tid >> 5;     // 0..7
  const int m0   = blockIdx.x * 128;
  const int n0   = blockIdx.y * 128;

  v8f acc[8];
#pragma unroll
  for (int i = 0; i < 8; ++i) acc[i] = (v8f){};

  for (int k0 = 0; k0 < K; k0 += 32) {
    {  // A tile 128x32: each thread 16 contiguous halfs (one b128 pair)
      int row = tid >> 1, col = (tid & 1) * 16;
      const bf16* src = A + (size_t)(m0 + row) * K + k0 + col;
      bf16* dst = As + row * 32 + col;
#pragma unroll
      for (int i = 0; i < 16; ++i) dst[i] = src[i];
    }
    {  // B tile 32x128 -> B^T in LDS. Each thread: 4k x 4n micro-tile,
       // written as four packed ds_store_b64 (contiguous along k in Bt).
      int kk = (tid >> 5) * 4;      // 0,4,...,28
      int nn = (tid & 31) * 4;      // 0,4,...,124
      const bf16* src = Bw + (size_t)(k0 + kk) * N + n0 + nn;
      bf16 hv[4][4];
#pragma unroll
      for (int i = 0; i < 4; ++i) {
        const bf16* s = src + (size_t)i * N;
#pragma unroll
        for (int j = 0; j < 4; ++j) hv[i][j] = s[j];
      }
#pragma unroll
      for (int j = 0; j < 4; ++j) {
        union { unsigned u32[2]; unsigned long long u64; } pk;
        pk.u32[0] = pack2(hv[0][j], hv[1][j]);
        pk.u32[1] = pack2(hv[2][j], hv[3][j]);
        *(unsigned long long*)(Bs + (nn + j) * 32 + kk) = pk.u64;
      }
    }
    __syncthreads();

    const int mrow = wv * 16 + (lane & 15);
    const int kb   = (lane < 16) ? 0 : 16;
    v16bf a = frag16(As + mrow * 32 + kb);
    const int kb2 = (lane < 16) ? 0 : 16;
#pragma unroll
    for (int nb = 0; nb < 8; ++nb) {
      int ncol = nb * 16 + (lane & 15);
      v16bf b = frag16(Bs + ncol * 32 + kb2);
      acc[nb] = wmma_bf16(a, b, acc[nb]);
    }
    __syncthreads();
  }

  // Epilogue. C layout: lane holds col n=lane&15; f[j] is row j + (lane>=16?8:0).
  const int rowoff = (lane & 16) ? 8 : 0;
#pragma unroll
  for (int nb = 0; nb < 8; ++nb) {
    int n = n0 + nb * 16 + (lane & 15);
    float bn = bias[n];
#pragma unroll
    for (int j = 0; j < 8; ++j) {
      int m = m0 + wv * 16 + rowoff + j;
      float v = acc[nb][j] + bn;
      if constexpr (EPI == 0) {
        outB[(size_t)m * ldo + n] = (bf16)v;
      } else if constexpr (EPI == 1) {
        // row m = (win, nidx); window-reverse + roll(+ss), add shortcut x.
        int N_tok = ws * ws, nWx = 64 / ws, nW = nWx * nWx;
        int win = m / N_tok, nidx = m % N_tok;
        int wx = win % nWx, wy = (win / nWx) % nWx, b = win / nW;
        int iy = nidx / ws, ix = nidx % ws;
        int h = (wy * ws + iy + ss) & 63;
        int w = (wx * ws + ix + ss) & 63;
        size_t sidx = ((size_t)(b * 4096 + h * 64 + w)) * 512 + cbase + n;
        outF[sidx] = resF[sidx] + v;
      } else if constexpr (EPI == 2) {
        float u = v + 0.044715f * v * v * v;
        float g = 0.5f * v * (1.0f + tanhf(0.7978845608028654f * u));
        outB[(size_t)m * ldo + n] = (bf16)g;
      } else {
        size_t idx = (size_t)m * ldo + n;
        outF[idx] = resF[idx] + v;
      }
    }
  }
}

// ---------------------------------------------------------------------------
// LN1 fused with roll+window-partition scatter into bf16 buffers per type.
// One block (128 threads) per token; C=512.
// ---------------------------------------------------------------------------
__device__ __forceinline__ size_t win_row(int b, int h, int w, int ws, int ss) {
  int hp = (h - ss) & 63, wp = (w - ss) & 63;
  int nWx = 64 / ws;
  int wy = hp / ws, iy = hp % ws, wx = wp / ws, ix = wp % ws;
  return ((size_t)((b * nWx + wy) * nWx + wx)) * (ws * ws) + iy * ws + ix;
}

__global__ void ln1_scatter_kernel(const float* __restrict__ x,
                                   const float* __restrict__ g,
                                   const float* __restrict__ be,
                                   bf16* __restrict__ xw0,
                                   bf16* __restrict__ xw1) {
  const int tok = blockIdx.x;   // [0, 65536)
  const int tid = threadIdx.x;  // 128
  const float* row = x + (size_t)tok * 512;
  float v[4], s = 0.f, sq = 0.f;
#pragma unroll
  for (int i = 0; i < 4; ++i) {
    v[i] = row[tid * 4 + i]; s += v[i]; sq += v[i] * v[i];
  }
  __shared__ float rs[128], rq[128];
  rs[tid] = s; rq[tid] = sq; __syncthreads();
  for (int o = 64; o > 0; o >>= 1) {
    if (tid < o) { rs[tid] += rs[tid + o]; rq[tid] += rq[tid + o]; }
    __syncthreads();
  }
  float mu  = rs[0] * (1.0f / 512.0f);
  float var = rq[0] * (1.0f / 512.0f) - mu * mu;
  float inv = rsqrtf(var + 1e-5f);

  int b = tok >> 12, hw = tok & 4095, h = hw >> 6, w = hw & 63;
  size_t r0 = win_row(b, h, w, 8, 4);
  size_t r1 = win_row(b, h, w, 16, 8);
  // two packed b32 stores instead of four b16
  bf16 y[4];
#pragma unroll
  for (int i = 0; i < 4; ++i) {
    int c = tid * 4 + i;
    y[i] = (bf16)((v[i] - mu) * inv * g[c] + be[c]);
  }
  int c0 = tid * 4;
  if (c0 < 256) {
    *(unsigned*)(xw0 + r0 * 256 + c0)     = pack2(y[0], y[1]);
    *(unsigned*)(xw0 + r0 * 256 + c0 + 2) = pack2(y[2], y[3]);
  } else {
    int cc = c0 - 256;
    *(unsigned*)(xw1 + r1 * 256 + cc)     = pack2(y[0], y[1]);
    *(unsigned*)(xw1 + r1 * 256 + cc + 2) = pack2(y[2], y[3]);
  }
}

// Generic LN: optional fp32 out (may alias input) and/or bf16 out.
__global__ void ln_kernel(const float* __restrict__ in,
                          const float* __restrict__ g,
                          const float* __restrict__ be,
                          float* outF, bf16* outB) {
  const int tok = blockIdx.x;
  const int tid = threadIdx.x;
  const float* row = in + (size_t)tok * 512;
  float v[4], s = 0.f, sq = 0.f;
#pragma unroll
  for (int i = 0; i < 4; ++i) {
    v[i] = row[tid * 4 + i]; s += v[i]; sq += v[i] * v[i];
  }
  __shared__ float rs[128], rq[128];
  rs[tid] = s; rq[tid] = sq; __syncthreads();
  for (int o = 64; o > 0; o >>= 1) {
    if (tid < o) { rs[tid] += rs[tid + o]; rq[tid] += rq[tid + o]; }
    __syncthreads();
  }
  float mu  = rs[0] * (1.0f / 512.0f);
  float var = rq[0] * (1.0f / 512.0f) - mu * mu;
  float inv = rsqrtf(var + 1e-5f);
#pragma unroll
  for (int i = 0; i < 4; ++i) {
    int c = tid * 4 + i;
    float y = (v[i] - mu) * inv * g[c] + be[c];
    if (outF) outF[(size_t)tok * 512 + c] = y;
    if (outB) outB[(size_t)tok * 512 + c] = (bf16)y;
  }
}

// ---------------------------------------------------------------------------
// Windowed attention, one block per (window, head, 64-query rowblock).
// 128 threads = 4 waves; each wave owns one 16-row tile. hd=32 => one WMMA
// K-step for QK^T; PV accumulates over N/32 key chunks.
// ---------------------------------------------------------------------------
__device__ __forceinline__ int regionf(int h, int ws, int ss) {
  return (h < 64 - ws) ? 0 : ((h < 64 - ss) ? 1 : 2);
}

template <int WS>
__global__ void attn_kernel(const bf16* __restrict__ qkv,
                            const float* __restrict__ rpb,
                            bf16* __restrict__ attn_out, int shift) {
  constexpr int N   = WS * WS;
  constexpr int RB  = N / 64;       // rowblocks per window
  constexpr int nWx = 64 / WS;
  constexpr int nW  = nWx * nWx;
  extern __shared__ char smem[];
  float* S  = (float*)smem;                             // 64 x N fp32 logits
  bf16*  P  = (bf16*)(smem + 64 * N * 4);               // 64 x N probs
  bf16*  Vt = (bf16*)(smem + 64 * N * 4 + 64 * N * 2);  // 32 x N  V^T

  const int tid = threadIdx.x, lane = tid & 31, wv = tid >> 5;
  const int id   = blockIdx.x;
  const int rb   = id % RB;
  const int head = (id / RB) % 8;
  const int winb = id / (RB * 8);   // [0, B*nW)
  const int win  = winb % nW;       // window within batch (for mask)
  const int R0   = rb * 64;

  const bf16* qbase = qkv + (size_t)winb * N * 768 + head * 32;
  const bf16* kbase = qbase + 256;
  const bf16* vbase = qbase + 512;

  // Stage V^T (32 x N) in LDS; pack key pairs -> b32 stores.
  for (int idx = tid; idx < 16 * N; idx += 128) {
    int d = idx & 31, kp = idx >> 5;  // key pair
    bf16 v0 = vbase[(size_t)(2 * kp) * 768 + d];
    bf16 v1 = vbase[(size_t)(2 * kp + 1) * 768 + d];
    *(unsigned*)(Vt + d * N + 2 * kp) = pack2(v0, v1);
  }

  // S = Q K^T (row tile per wave, all key tiles).
  {
    int m  = R0 + wv * 16 + (lane & 15);
    int kb = (lane < 16) ? 0 : 16;
    const bf16* qp = qbase + (size_t)m * 768 + kb;
    v16bf a = frag16(qp);
    int kb2 = (lane < 16) ? 0 : 16;
    int rloc = wv * 16 + ((lane & 16) ? 8 : 0);
#pragma unroll 2
    for (int kt = 0; kt < N / 16; ++kt) {
      int key = kt * 16 + (lane & 15);
      v16bf b = frag16(kbase + (size_t)key * 768 + kb2);
      v8f c = (v8f){};
      c = wmma_bf16(a, b, c);
      int col = kt * 16 + (lane & 15);
#pragma unroll
      for (int j = 0; j < 8; ++j) S[(rloc + j) * N + col] = c[j];
    }
  }
  __syncthreads();

  // Softmax with relative-position bias + shifted-window mask.
  const float scale = 0.17677669529663687f;  // 32^-0.5
  for (int r = tid; r < 64; r += 128) {
    int q  = R0 + r;
    int qy = q / WS, qx = q % WS;
    int hq = (win / nWx) * WS + qy;
    int wq = (win % nWx) * WS + qx;
    int regq = 3 * regionf(hq, WS, shift) + regionf(wq, WS, shift);
    float* Srow = S + r * N;
    float mx = -1e30f;
    for (int j = 0; j < N; ++j) {
      int ky = j / WS, kx = j % WS;
      int rel = (qy - ky + WS - 1) * (2 * WS - 1) + (qx - kx + WS - 1);
      float bias = rpb[rel * 8 + head];
      int hk = (win / nWx) * WS + ky;
      int wk = (win % nWx) * WS + kx;
      int regk = 3 * regionf(hk, WS, shift) + regionf(wk, WS, shift);
      float msk = (regq != regk) ? -100.0f : 0.0f;
      float val = Srow[j] * scale + bias + msk;
      Srow[j] = val;
      mx = fmaxf(mx, val);
    }
    float sum = 0.f;
    for (int j = 0; j < N; ++j) {
      float e = __expf(Srow[j] - mx);
      Srow[j] = e; sum += e;
    }
    float inv = 1.0f / sum;
    bf16* Prow = P + r * N;
    for (int j = 0; j < N; j += 2) {
      *(unsigned*)(Prow + j) =
          pack2((bf16)(Srow[j] * inv), (bf16)(Srow[j + 1] * inv));
    }
  }
  __syncthreads();

  // O = P V : 16x32 output tile per wave, accumulate over N/32 key chunks.
  {
    int mloc = wv * 16 + (lane & 15);
    int rloc = wv * 16 + ((lane & 16) ? 8 : 0);
#pragma unroll
    for (int dt = 0; dt < 2; ++dt) {
      v8f acc = (v8f){};
      for (int kt = 0; kt < N / 32; ++kt) {
        int kb = kt * 32 + ((lane < 16) ? 0 : 16);
        v16bf a = frag16(P + mloc * N + kb);
        int d   = dt * 16 + (lane & 15);
        int kb2 = kt * 32 + ((lane < 16) ? 0 : 16);
        v16bf b = frag16(Vt + d * N + kb2);
        acc = wmma_bf16(a, b, acc);
      }
      int dcol = dt * 16 + (lane & 15);
#pragma unroll
      for (int j = 0; j < 8; ++j) {
        int q = R0 + rloc + j;
        attn_out[((size_t)winb * N + q) * 256 + head * 32 + dcol] = (bf16)acc[j];
      }
    }
  }
}

// ---------------------------------------------------------------------------
// Depthwise conv (SAME, odd k) on one 256-channel half; writes x3 = x2n + conv
// into out (d_out). One block per pixel, 256 threads over channels.
// ---------------------------------------------------------------------------
__global__ void dwconv_kernel(const float* __restrict__ x2n,
                              const float* __restrict__ wgt,
                              const float* __restrict__ bias,
                              float* __restrict__ out, int k, int cbase) {
  const int pix = blockIdx.x;           // b*4096 + h*64 + w
  const int c   = threadIdx.x;          // 256
  const int h = (pix >> 6) & 63, w = pix & 63;
  const int base = pix - (h << 6) - w;  // b*4096
  const int pad = k >> 1;
  float acc = bias[c];
  for (int dy = 0; dy < k; ++dy) {
    int hh = h + dy - pad;
    if (hh < 0 || hh >= 64) continue;
    for (int dx = 0; dx < k; ++dx) {
      int ww = w + dx - pad;
      if (ww < 0 || ww >= 64) continue;
      acc += x2n[((size_t)(base + (hh << 6) + ww)) * 512 + cbase + c] *
             wgt[(dy * k + dx) * 256 + c];
    }
  }
  size_t idx = (size_t)pix * 512 + cbase + c;
  out[idx] = x2n[idx] + acc;
}

__global__ void f2bf_kernel(const float* __restrict__ in, bf16* __restrict__ out,
                            int n) {
  int i = blockIdx.x * 256 + threadIdx.x;
  if (i < n) out[i] = (bf16)in[i];
}

// ---------------------------------------------------------------------------
// Workspace layout (bytes), with lifetime-based overlap. Peak ~475 MB.
// ---------------------------------------------------------------------------
static constexpr size_t XW_B   = (size_t)65536 * 256 * 2;   //  33,554,432
static constexpr size_t QKV_B  = (size_t)65536 * 768 * 2;   // 100,663,296
static constexpr size_t ATTN_B = XW_B;
static constexpr size_t X2_B   = (size_t)65536 * 512 * 4;   // 134,217,728

static constexpr size_t OFF_XW0   = 0;
static constexpr size_t OFF_XW1   = OFF_XW0 + XW_B;
static constexpr size_t OFF_QKV0  = OFF_XW1 + XW_B;
static constexpr size_t OFF_QKV1  = OFF_QKV0 + QKV_B;
static constexpr size_t OFF_ATTN0 = OFF_QKV1 + QKV_B;        // 268,435,456
static constexpr size_t OFF_ATTN1 = OFF_ATTN0 + ATTN_B;
static constexpr size_t OFF_M1    = 0;                       // reuses xw+qkv
static constexpr size_t OFF_H3    = OFF_ATTN0;               // reuses attn
static constexpr size_t OFF_X2    = OFF_ATTN1 + ATTN_B;      // 335,544,320
static constexpr size_t OFF_WB    = OFF_X2 + X2_B;           // 469,762,048
static constexpr size_t OFF_WQKV0 = OFF_WB;
static constexpr size_t OFF_WQKV1 = OFF_WQKV0 + (size_t)196608 * 2;
static constexpr size_t OFF_WPRJ0 = OFF_WQKV1 + (size_t)196608 * 2;
static constexpr size_t OFF_WPRJ1 = OFF_WPRJ0 + (size_t)65536 * 2;
static constexpr size_t OFF_WMLP1 = OFF_WPRJ1 + (size_t)65536 * 2;
static constexpr size_t OFF_WMLP2 = OFF_WMLP1 + (size_t)1048576 * 2;

extern "C" void kernel_launch(void* const* d_in, const int* in_sizes, int n_in,
                              void* d_out, int out_size, void* d_ws,
                              size_t ws_size, hipStream_t stream) {
  (void)in_sizes; (void)n_in; (void)out_size; (void)ws_size;
  const float* x   = (const float*)d_in[0];
  const float* g1  = (const float*)d_in[1];
  const float* g2  = (const float*)d_in[2];
  const float* g3  = (const float*)d_in[3];
  const float* be1 = (const float*)d_in[4];
  const float* be2 = (const float*)d_in[5];
  const float* be3 = (const float*)d_in[6];
  const float* qkv_w[2]  = {(const float*)d_in[7],  (const float*)d_in[14]};
  const float* qkv_b[2]  = {(const float*)d_in[8],  (const float*)d_in[15]};
  const float* proj_w[2] = {(const float*)d_in[9],  (const float*)d_in[16]};
  const float* proj_b[2] = {(const float*)d_in[10], (const float*)d_in[17]};
  const float* rpb[2]    = {(const float*)d_in[11], (const float*)d_in[18]};
  const float* conv_w[2] = {(const float*)d_in[12], (const float*)d_in[19]};
  const float* conv_b[2] = {(const float*)d_in[13], (const float*)d_in[20]};
  const float* mlp_w1 = (const float*)d_in[21];
  const float* mlp_b1 = (const float*)d_in[22];
  const float* mlp_w2 = (const float*)d_in[23];
  const float* mlp_b2 = (const float*)d_in[24];
  float* out = (float*)d_out;
  char*  ws  = (char*)d_ws;

  bf16* xw[2]   = {(bf16*)(ws + OFF_XW0),  (bf16*)(ws + OFF_XW1)};
  bf16* qkv[2]  = {(bf16*)(ws + OFF_QKV0), (bf16*)(ws + OFF_QKV1)};
  bf16* attn[2] = {(bf16*)(ws + OFF_ATTN0),(bf16*)(ws + OFF_ATTN1)};
  bf16* h3      = (bf16*)(ws + OFF_H3);
  bf16* m1      = (bf16*)(ws + OFF_M1);
  float* x2     = (float*)(ws + OFF_X2);
  bf16* wqkv[2] = {(bf16*)(ws + OFF_WQKV0), (bf16*)(ws + OFF_WQKV1)};
  bf16* wprj[2] = {(bf16*)(ws + OFF_WPRJ0), (bf16*)(ws + OFF_WPRJ1)};
  bf16* wmlp1   = (bf16*)(ws + OFF_WMLP1);
  bf16* wmlp2   = (bf16*)(ws + OFF_WMLP2);

  // 0) weight fp32 -> bf16
  for (int t = 0; t < 2; ++t) {
    f2bf_kernel<<<(196608 + 255) / 256, 256, 0, stream>>>(qkv_w[t], wqkv[t], 196608);
    f2bf_kernel<<<(65536 + 255) / 256, 256, 0, stream>>>(proj_w[t], wprj[t], 65536);
  }
  f2bf_kernel<<<(1048576 + 255) / 256, 256, 0, stream>>>(mlp_w1, wmlp1, 1048576);
  f2bf_kernel<<<(1048576 + 255) / 256, 256, 0, stream>>>(mlp_w2, wmlp2, 1048576);

  // 1) LN1 + roll + window-partition -> xw[t] (bf16)
  ln1_scatter_kernel<<<65536, 128, 0, stream>>>(x, g1, be1, xw[0], xw[1]);

  // 2) QKV GEMMs: (65536 x 256) @ (256 x 768) -> bf16 qkv[t]
  for (int t = 0; t < 2; ++t) {
    gemm_bf16_kernel<0><<<dim3(512, 6), 256, 0, stream>>>(
        xw[t], wqkv[t], qkv_b[t], 65536, 768, 256,
        nullptr, qkv[t], 768, nullptr, 0, 0, 0);
  }

  // 3) Attention (both types launch 8192 blocks of 128 threads)
  attn_kernel<8><<<8192, 128, 64 * 64 * 4 + 64 * 64 * 2 + 32 * 64 * 2, stream>>>(
      qkv[0], rpb[0], attn[0], 4);
  attn_kernel<16><<<8192, 128, 64 * 256 * 4 + 64 * 256 * 2 + 32 * 256 * 2, stream>>>(
      qkv[1], rpb[1], attn[1], 8);

  // 4) proj GEMMs with window-reverse + roll + shortcut residual -> x2 (fp32)
  for (int t = 0; t < 2; ++t) {
    gemm_bf16_kernel<1><<<dim3(512, 2), 256, 0, stream>>>(
        attn[t], wprj[t], proj_b[t], 65536, 256, 256,
        x2, nullptr, 256, x, t ? 16 : 8, t ? 8 : 4, t * 256);
  }

  // 5) LN2 in place (each block fully owns its row)
  ln_kernel<<<65536, 128, 0, stream>>>(x2, g2, be2, x2, nullptr);

  // 6) depthwise convs; x3 = x2n + conv -> d_out
  dwconv_kernel<<<65536, 256, 0, stream>>>(x2, conv_w[0], conv_b[0], out, 9, 0);
  dwconv_kernel<<<65536, 256, 0, stream>>>(x2, conv_w[1], conv_b[1], out, 17, 256);

  // 7) LN3: h3 (bf16) from x3 (= d_out)
  ln_kernel<<<65536, 128, 0, stream>>>(out, g3, be3, nullptr, h3);

  // 8) MLP fc1 + GELU -> m1 (bf16)
  gemm_bf16_kernel<2><<<dim3(512, 16), 256, 0, stream>>>(
      h3, wmlp1, mlp_b1, 65536, 2048, 512,
      nullptr, m1, 2048, nullptr, 0, 0, 0);

  // 9) MLP fc2 + residual(x3) -> d_out (in place: read x3, write result)
  gemm_bf16_kernel<3><<<dim3(512, 4), 256, 0, stream>>>(
      m1, wmlp2, mlp_b2, 65536, 512, 2048,
      out, nullptr, 512, out, 0, 0, 0);
}
